// LTEAttention_70093866271294
// MI455X (gfx1250) — matrix operand order repaired and checked by reading, not compiled, not run on hardware
//
#include <hip/hip_runtime.h>

// ---------------------------------------------------------------------------
// Problem constants (match reference)
// ---------------------------------------------------------------------------
#define BB    2
#define LL    2048
#define DD    2048
#define HH    16
#define HKV   4
#define HD    128
#define WIN   512
#define SINKN 4
#define NQKV  3072           // q(2048) | k(512) | v(512)

typedef __attribute__((ext_vector_type(16))) __bf16 bf16x16;
typedef __attribute__((ext_vector_type(8)))  float  v8f;

union FragU {
  unsigned u[8];
  bf16x16  v;
};

__device__ __forceinline__ v8f v8zero() {
  v8f z = {0.f, 0.f, 0.f, 0.f, 0.f, 0.f, 0.f, 0.f};
  return z;
}

__device__ __forceinline__ unsigned short f2bf(float f) {
  unsigned u = __float_as_uint(f);
  u = (u + 0x7fffu + ((u >> 16) & 1u)) >> 16;   // round-to-nearest-even
  return (unsigned short)u;
}
__device__ __forceinline__ unsigned pack2bf(float lo, float hi) {
  return (unsigned)f2bf(lo) | ((unsigned)f2bf(hi) << 16);
}
__device__ __forceinline__ v8f wmma_bf16x(const FragU& a, const FragU& b, v8f c) {
  // D = A(16x32 bf16) * B(32x16 bf16) + C(16x16 f32)
  return __builtin_amdgcn_wmma_f32_16x16x32_bf16(false, a.v, false, b.v,
                                                 (short)0, c, false, false);
}

// ---------------------------------------------------------------------------
// fp32 -> bf16 elementwise convert
// ---------------------------------------------------------------------------
__global__ __launch_bounds__(256) void k_cvt_bf16(const float* __restrict__ in,
                                                  unsigned short* __restrict__ out) {
  int idx = blockIdx.x * 256 + threadIdx.x;
  out[idx] = f2bf(in[idx]);
}

// Pack Wq|Wk|Wv -> bf16 [2048][3072] row-major (K x N)
__global__ __launch_bounds__(256) void k_pack_wqkv(const float* __restrict__ Wq,
                                                   const float* __restrict__ Wk,
                                                   const float* __restrict__ Wv,
                                                   unsigned short* __restrict__ out) {
  int idx = blockIdx.x * 256 + threadIdx.x;      // 2048*3072
  int n = idx % NQKV;
  int k = idx / NQKV;
  float val;
  if (n < 2048)       val = Wq[k * 2048 + n];
  else if (n < 2560)  val = Wk[k * 512 + (n - 2048)];
  else                val = Wv[k * 512 + (n - 2560)];
  out[idx] = f2bf(val);
}

// ---------------------------------------------------------------------------
// bf16 GEMM: C[f32, MxN] = A[bf16, MxK, rm] * B[bf16, KxN, rm]
// Compile-time N,K -> immediate-offset epilogue stores.
// Block = 256 threads (8 waves). Tile 128x128, K-step 32.
// Double-buffered LDS; next tile's global loads issued before current tile's
// compute; global_prefetch for the K+2 tile.
// ---------------------------------------------------------------------------
template <int N, int K>
__global__ __launch_bounds__(256) void k_gemm_bf16(const unsigned short* __restrict__ A,
                                                   const unsigned short* __restrict__ Bm,
                                                   float* __restrict__ C) {
  __shared__ unsigned AsU[2][128 * 18];          // [buf][row][36 halves], padded
  __shared__ unsigned BsU[2][128 * 18];          // [buf][col][36 halves] (k-minor)

  const int t    = threadIdx.x;
  const int lane = t & 31, wvi = t >> 5;
  const int wm   = wvi >> 1, wn = wvi & 1;
  const int hf   = lane >> 4, l16 = lane & 15;
  const size_t m0 = (size_t)blockIdx.y * 128;
  const size_t n0 = (size_t)blockIdx.x * 128;

  v8f acc[2][4];
  for (int i = 0; i < 2; i++)
    for (int j = 0; j < 4; j++) acc[i][j] = v8zero();

  const int r = t >> 1, seg = t & 1;             // A fill: row, 16-half segment
  const int kk = t >> 3, nseg = (t & 7) * 16;    // B fill: k row, 16-col segment

  const unsigned short* aG = A + (m0 + r) * K + seg * 16;
  const unsigned short* bG = Bm + (size_t)kk * N + n0 + nseg;

  auto loadA = [&](int kt, uint4& d0, uint4& d1) {
    const uint4* gp = (const uint4*)(aG + kt);
    d0 = gp[0]; d1 = gp[1];
  };
  auto loadB = [&](int kt, uint4& d0, uint4& d1) {
    const uint4* gp = (const uint4*)(bG + (size_t)kt * N);
    d0 = gp[0]; d1 = gp[1];
  };
  auto storeTile = [&](int buf, uint4 a0, uint4 a1, uint4 b0, uint4 b1) {
    unsigned* ad = &AsU[buf][r * 18 + seg * 8];
    ad[0] = a0.x; ad[1] = a0.y; ad[2] = a0.z; ad[3] = a0.w;
    ad[4] = a1.x; ad[5] = a1.y; ad[6] = a1.z; ad[7] = a1.w;
    unsigned short* bs = (unsigned short*)(&BsU[buf][0]);
    unsigned w0[4] = {b0.x, b0.y, b0.z, b0.w};
    unsigned w1[4] = {b1.x, b1.y, b1.z, b1.w};
#pragma unroll
    for (int j = 0; j < 8; j++)
      bs[(nseg + j) * 36 + kk] =
          (unsigned short)((w0[j >> 1] >> ((j & 1) * 16)) & 0xffffu);
#pragma unroll
    for (int j = 0; j < 8; j++)
      bs[(nseg + 8 + j) * 36 + kk] =
          (unsigned short)((w1[j >> 1] >> ((j & 1) * 16)) & 0xffffu);
  };
  auto compute = [&](int buf) {
    FragU af[2], bfr[4];
#pragma unroll
    for (int sm = 0; sm < 2; sm++) {
      int row = 32 * wm + 16 * sm + l16;
#pragma unroll
      for (int v = 0; v < 8; v++) {
        // 16-bit A layout: v<4 -> K = 2v + 8*half ; v>=4 -> 16 + 2(v-4) + 8*half
        int k0 = (v < 4) ? (2 * v + 8 * hf) : (16 + 2 * (v - 4) + 8 * hf);
        af[sm].u[v] = AsU[buf][row * 18 + (k0 >> 1)];
      }
    }
#pragma unroll
    for (int sn = 0; sn < 4; sn++) {
      int col = 64 * wn + 16 * sn + l16;
#pragma unroll
      for (int v = 0; v < 8; v++) {
        int k0 = (v < 4) ? (2 * v + 8 * hf) : (16 + 2 * (v - 4) + 8 * hf);
        bfr[sn].u[v] = BsU[buf][col * 18 + (k0 >> 1)];
      }
    }
#pragma unroll
    for (int sm = 0; sm < 2; sm++)
#pragma unroll
      for (int sn = 0; sn < 4; sn++)
        acc[sm][sn] = wmma_bf16x(af[sm], bfr[sn], acc[sm][sn]);
  };

  constexpr int NT = K / 32;
  {
    uint4 a0, a1, b0, b1;
    loadA(0, a0, a1);
    loadB(0, b0, b1);
    storeTile(0, a0, a1, b0, b1);
  }
  __syncthreads();
  for (int i = 1; i < NT; i++) {
    const int kt = i * 32;
    uint4 a0, a1, b0, b1;
    loadA(kt, a0, a1);                 // next tile: overlap with compute below
    loadB(kt, b0, b1);
    if (i + 1 < NT) {                  // warm L2/WGP$ two tiles ahead
      __builtin_prefetch(aG + kt + 32, 0, 3);
      __builtin_prefetch(bG + (size_t)(kt + 32) * N, 0, 3);
    }
    compute((i - 1) & 1);
    storeTile(i & 1, a0, a1, b0, b1);
    __syncthreads();
  }
  compute((NT - 1) & 1);

  // epilogue: one base per tile, immediate row offsets (v*N folds into offset)
#pragma unroll
  for (int sm = 0; sm < 2; sm++)
#pragma unroll
    for (int sn = 0; sn < 4; sn++) {
      float* cp = C + (m0 + 32 * wm + 16 * sm + 8 * hf) * N +
                  (n0 + 64 * wn + 16 * sn + l16);
#pragma unroll
      for (int v = 0; v < 8; v++) cp[(size_t)v * N] = acc[sm][sn][v];
    }
}

// ---------------------------------------------------------------------------
// RoPE + layout staging:
//  qkv f32 [B,L,3072]  ->  q bf16 [B,H,L,HD], k bf16 [B,HKV,L,HD],
//                          v^T bf16 [B,HKV,HD,L], conv input f32 [B,1024,L]
// ---------------------------------------------------------------------------
__global__ __launch_bounds__(256) void k_rope(const float* __restrict__ qkv,
                                              unsigned short* __restrict__ qb,
                                              unsigned short* __restrict__ kbuf,
                                              unsigned short* __restrict__ vtb,
                                              float* __restrict__ convin) {
  const int bl = blockIdx.x;                 // b*L + l
  const int b  = bl >> 11;
  const int l  = bl & 2047;
  const float lf = (float)l;
  const float* row = qkv + (size_t)bl * NQKV;

  // Q heads
  for (int idx = threadIdx.x; idx < 1024; idx += 256) {
    int hq = idx >> 6, d = idx & 63;
    float invf = __expf(-(float)d * 0.14391157f);   // ln(1e4)/64
    float sn, cs; __sincosf(lf * invf, &sn, &cs);
    float x0 = row[hq * 128 + d], x1 = row[hq * 128 + d + 64];
    unsigned short* dst = qb + (((size_t)(b * HH + hq)) * LL + l) * HD;
    dst[d]      = f2bf(x0 * cs - x1 * sn);
    dst[d + 64] = f2bf(x1 * cs + x0 * sn);
  }
  // K heads (+ router conv input, channels hk*256 + [0,128))
  {
    int idx = threadIdx.x;
    int hk = idx >> 6, d = idx & 63;
    float invf = __expf(-(float)d * 0.14391157f);
    float sn, cs; __sincosf(lf * invf, &sn, &cs);
    float x0 = row[2048 + hk * 128 + d], x1 = row[2048 + hk * 128 + d + 64];
    float r0 = x0 * cs - x1 * sn;
    float r1 = x1 * cs + x0 * sn;
    unsigned short* dst = kbuf + (((size_t)(b * HKV + hk)) * LL + l) * HD;
    dst[d]      = f2bf(r0);
    dst[d + 64] = f2bf(r1);
    float* cv = convin + ((size_t)b * 1024 + hk * 256) * LL + l;
    cv[(size_t)d * LL]        = r0;
    cv[(size_t)(d + 64) * LL] = r1;
  }
  // V heads: store transposed [hkv][hd][L] (+ conv channels hk*256+128+d)
  for (int idx = threadIdx.x; idx < 512; idx += 256) {
    int hk = idx >> 7, d = idx & 127;
    float val = row[2560 + idx];
    vtb[(((size_t)(b * HKV + hk)) * HD + d) * LL + l] = f2bf(val);
    convin[((size_t)b * 1024 + hk * 256 + 128 + d) * LL + l] = val;
  }
}

// ---------------------------------------------------------------------------
// Grouped conv (groups=4, k=3, SAME) + optional SiLU. Layouts [B,C,L] f32.
// ---------------------------------------------------------------------------
__global__ __launch_bounds__(256) void k_conv3(const float* __restrict__ in,
                                               const float* __restrict__ w,
                                               const float* __restrict__ bias,
                                               float* __restrict__ out,
                                               int Cout, int CinG, int doSilu) {
  int idx = blockIdx.x * 256 + threadIdx.x;   // (b*Cout + co)*L + l
  int l  = idx & 2047;
  int co = (idx >> 11) % Cout;
  int b  = idx / (Cout << 11);
  int g  = co / (Cout >> 2);
  const float* ip = in + (((size_t)b * (CinG * 4)) + (size_t)g * CinG) * LL + l;
  const float* wp = w + (size_t)co * CinG * 3;
  float acc = bias[co];
  for (int ci = 0; ci < CinG; ci++) {
    const float* rr = ip + (size_t)ci * LL;
    float x0 = (l > 0)    ? rr[-1] : 0.f;
    float x1 = rr[0];
    float x2 = (l < 2047) ? rr[1]  : 0.f;
    acc += wp[3 * ci] * x0 + wp[3 * ci + 1] * x1 + wp[3 * ci + 2] * x2;
  }
  if (doSilu) acc = acc / (1.f + __expf(-acc));
  out[idx] = acc;
}

// 1x1 grouped proj (bias 0.5) -> selected mask (relu(x)>0 ⇔ x>0; force sinks)
__global__ __launch_bounds__(256) void k_proj_sel(const float* __restrict__ in,
                                                  const float* __restrict__ pw,
                                                  const float* __restrict__ pb,
                                                  unsigned char* __restrict__ sel) {
  int idx = blockIdx.x * 256 + threadIdx.x;   // [B][HKV][L]
  int l  = idx & 2047;
  int hk = (idx >> 11) & 3;
  int b  = idx >> 13;
  const float* ip = in + ((size_t)b * 128 + hk * 32) * LL + l;
  float acc = pb[hk];
  for (int ci = 0; ci < 32; ci++) acc += pw[hk * 32 + ci] * ip[(size_t)ci * LL];
  sel[idx] = (acc > 0.f || l < SINKN) ? 1 : 0;
}

// selection bitmask per 32-key block: bit i = selected[block*32 + i]
// (smask != 0 also serves as the block-any skip test; sinks force block 0 on)
__global__ __launch_bounds__(256) void k_selmask(const unsigned char* __restrict__ sel,
                                                 unsigned* __restrict__ sm) {
  int idx = blockIdx.x * 256 + threadIdx.x;
  if (idx >= BB * HKV * (LL / 32)) return;
  const unsigned char* s = sel + (size_t)idx * 32;
  unsigned m = 0;
#pragma unroll
  for (int i = 0; i < 32; i++) m |= (s[i] ? 1u : 0u) << i;
  sm[idx] = m;
}

// ---------------------------------------------------------------------------
// Attention: one wave per 16-query tile, flash-style over 32-key chunks.
// Computes S^T = K·Q^T and O^T = V^T·P^T so softmax reductions are in-lane
// (+ one shfl_xor(16)) and all WMMA operand loads are contiguous uint4.
// Selection handled as a per-block u32 bitmask -> fully branchless masking,
// and the same word drives the block-skip test.
// ---------------------------------------------------------------------------
__global__ __launch_bounds__(256) void k_attn(const unsigned short* __restrict__ qb,
                                              const unsigned short* __restrict__ kbuf,
                                              const unsigned short* __restrict__ vtb,
                                              const unsigned* __restrict__ selmask,
                                              unsigned short* __restrict__ attnout) {
  const int lane = threadIdx.x & 31;
  const int wv   = threadIdx.x >> 5;
  const int gid  = blockIdx.x * 8 + wv;       // b*H*(L/16) waves total = 4096
  const int qt   = gid & 127;
  const int h    = (gid >> 7) & 15;
  const int b    = gid >> 11;
  const int hkv  = h >> 2;                    // G = H/HKV = 4
  const int hf   = lane >> 4, l16 = lane & 15;
  const int qbase = qt << 4;
  const float SCALE = 0.0883883476f;          // 1/sqrt(128)

  // Q as B-operand fragments (Q^T: 32 hd x 16 q), 4 HD chunks, loaded once
  FragU qf[4];
  {
    const unsigned short* qrow =
        qb + (((size_t)(b * HH + h)) * LL + (qbase + l16)) * HD;
    for (int c = 0; c < 4; c++) {
      const uint4* p = (const uint4*)(qrow + 32 * c + 16 * hf);
      uint4 d0 = p[0], d1 = p[1];
      qf[c].u[0] = d0.x; qf[c].u[1] = d0.y; qf[c].u[2] = d0.z; qf[c].u[3] = d0.w;
      qf[c].u[4] = d1.x; qf[c].u[5] = d1.y; qf[c].u[6] = d1.z; qf[c].u[7] = d1.w;
    }
  }

  v8f o[8];                                   // O^T accum: 8 hd-subtiles x 16 q
  for (int i = 0; i < 8; i++) o[i] = v8zero();
  float mprev = -1e30f, lsum = 0.f;

  const unsigned short* kp = kbuf + ((size_t)(b * HKV + hkv)) * LL * HD;
  const unsigned short* vp = vtb + ((size_t)(b * HKV + hkv)) * HD * LL;
  const unsigned* smp = selmask + (size_t)(b * HKV + hkv) * (LL / 32);
  const int iq = qbase + l16;                 // this lane's query index

  for (int kb = 0; kb <= qbase + 15; kb += 32) {
    const unsigned smask = smp[kb >> 5];
    // skip: entirely outside sliding window and no selected key in block
    // (block 0 always has smask != 0 because of the forced sinks)
    if (((qbase - (kb + 31)) >= WIN) && (smask == 0u)) continue;

    // warm the cache for the next key block while we compute this one
    if (kb + 32 <= qbase + 15) {
      __builtin_prefetch(kp + (size_t)(kb + 32 + l16) * HD, 0, 3);
      __builtin_prefetch(vp + (size_t)(16 * (lane >> 2) + (lane & 3)) * LL + kb + 32, 0, 3);
    }

    float p0a[8], p1a[8];
    for (int tile = 0; tile < 2; tile++) {    // two 16-key S^T tiles
      v8f s = v8zero();
      const unsigned short* krow = kp + (size_t)(kb + 16 * tile + l16) * HD;
      for (int c = 0; c < 4; c++) {
        FragU af;                             // K as A-operand (16 keys x 32 hd)
        uint4 d0 = *(const uint4*)(krow + 32 * c + 16 * hf);
        uint4 d1 = *(const uint4*)(krow + 32 * c + 16 * hf + 8);
        af.u[0] = d0.x; af.u[1] = d0.y; af.u[2] = d0.z; af.u[3] = d0.w;
        af.u[4] = d1.x; af.u[5] = d1.y; af.u[6] = d1.z; af.u[7] = d1.w;
        s = wmma_bf16x(af, qf[c], s);
      }
      float* dst = tile ? p1a : p0a;
#pragma unroll
      for (int v = 0; v < 8; v++) {           // S^T rows = keys v + 8*half
        const int rel = 16 * tile + v + 8 * hf;     // j & 31
        const int dj  = iq - (kb + rel);
        const bool okw = ((unsigned)dj < (unsigned)WIN) | ((smask >> rel) & 1u);
        const bool ok  = (dj >= 0) & okw;           // branchless, no loads
        dst[v] = ok ? s[v] * SCALE : -1e30f;
      }
    }

    // online softmax over keys (per query = per lane column).
    // Masked entries hold -1e30; after block 0 (always processed, key 0
    // always allowed) mnew is finite, so __expf underflows them to 0.
    float mx = p0a[0];
    for (int v = 1; v < 8; v++) mx = fmaxf(mx, p0a[v]);
    for (int v = 0; v < 8; v++) mx = fmaxf(mx, p1a[v]);
    mx = fmaxf(mx, __shfl_xor(mx, 16, 32));
    float mnew = fmaxf(mprev, mx);
    float corr = __expf(mprev - mnew);
    float ssum = 0.f;
#pragma unroll
    for (int v = 0; v < 8; v++) {
      p0a[v] = __expf(p0a[v] - mnew);
      p1a[v] = __expf(p1a[v] - mnew);
      ssum += p0a[v] + p1a[v];
    }
    ssum += __shfl_xor(ssum, 16, 32);
    lsum = lsum * corr + ssum;
    mprev = mnew;

    // build P^T B-fragment (32 keys x 16 q): half-swap keys via shfl_xor(16)
    float lowv[8], highv[8];
#pragma unroll
    for (int v = 0; v < 8; v++) {
      float other = hf ? p0a[v] : p1a[v];
      float sh = __shfl_xor(other, 16, 32);
      lowv[v]  = hf ? sh     : p0a[v];        // keys 0..7 of this half's tile
      highv[v] = hf ? p1a[v] : sh;            // keys 8..15
    }
    FragU pf;
    pf.u[0] = pack2bf(lowv[0],  lowv[1]);
    pf.u[1] = pack2bf(lowv[2],  lowv[3]);
    pf.u[2] = pack2bf(lowv[4],  lowv[5]);
    pf.u[3] = pack2bf(lowv[6],  lowv[7]);
    pf.u[4] = pack2bf(highv[0], highv[1]);
    pf.u[5] = pack2bf(highv[2], highv[3]);
    pf.u[6] = pack2bf(highv[4], highv[5]);
    pf.u[7] = pack2bf(highv[6], highv[7]);

    // O^T += V^T(16 hd x 32 keys) · P^T
    for (int s8 = 0; s8 < 8; s8++) {
      for (int v = 0; v < 8; v++) o[s8][v] *= corr;
      FragU vf;
      const unsigned short* vrow = vp + (size_t)(16 * s8 + l16) * LL + kb;
      uint4 d0 = *(const uint4*)(vrow + 16 * hf);
      uint4 d1 = *(const uint4*)(vrow + 16 * hf + 8);
      vf.u[0] = d0.x; vf.u[1] = d0.y; vf.u[2] = d0.z; vf.u[3] = d0.w;
      vf.u[4] = d1.x; vf.u[5] = d1.y; vf.u[6] = d1.z; vf.u[7] = d1.w;
      o[s8] = wmma_bf16x(vf, pf, o[s8]);
    }
  }

  // normalize and store attn output bf16 [B*L][H*HD] (A matrix of out proj)
  float inv = 1.f / lsum;
  unsigned short* orow =
      attnout + ((size_t)b * LL + qbase + l16) * (HH * HD) + h * HD;
  for (int s8 = 0; s8 < 8; s8++)
    for (int v = 0; v < 8; v += 2)
      *(unsigned*)(orow + 16 * s8 + 8 * hf + v) =
          pack2bf(o[s8][v] * inv, o[s8][v + 1] * inv);
}

// ---------------------------------------------------------------------------
// Host orchestration
// ---------------------------------------------------------------------------
extern "C" void kernel_launch(void* const* d_in, const int* in_sizes, int n_in,
                              void* d_out, int out_size, void* d_ws, size_t ws_size,
                              hipStream_t stream) {
  (void)in_sizes; (void)n_in; (void)out_size; (void)ws_size;
  const float* hidden = (const float*)d_in[0];
  const float* Wq  = (const float*)d_in[1];
  const float* Wk  = (const float*)d_in[2];
  const float* Wv  = (const float*)d_in[3];
  const float* Wo  = (const float*)d_in[4];
  const float* cw0 = (const float*)d_in[5];
  const float* cb0 = (const float*)d_in[6];
  const float* cw1 = (const float*)d_in[7];
  const float* cb1 = (const float*)d_in[8];
  const float* cw2 = (const float*)d_in[9];
  const float* cb2 = (const float*)d_in[10];
  const float* pw  = (const float*)d_in[11];
  const float* pb  = (const float*)d_in[12];
  float* out = (float*)d_out;

  char* base = (char*)d_ws;
  size_t off = 0;
  auto alloc = [&](size_t bytes) -> char* {
    char* p = base + off;
    off = (off + bytes + 255) & ~(size_t)255;
    return p;
  };
  float*          qkv    = (float*)alloc((size_t)4096 * 3072 * 4);       // 48 MB
  float*          convin = (float*)alloc((size_t)2 * 1024 * 2048 * 4);   // 16 MB
  unsigned short* hsb    = (unsigned short*)alloc((size_t)4096 * 2048 * 2);
  unsigned short* wqkvb  = (unsigned short*)alloc((size_t)2048 * 3072 * 2);
  unsigned short* wob    = (unsigned short*)alloc((size_t)2048 * 2048 * 2);
  unsigned short* qbb    = (unsigned short*)alloc((size_t)2 * 16 * 2048 * 128 * 2);
  unsigned short* kbb    = (unsigned short*)alloc((size_t)2 * 4 * 2048 * 128 * 2);
  unsigned short* vtb    = (unsigned short*)alloc((size_t)2 * 4 * 128 * 2048 * 2);
  unsigned char*  selb   = (unsigned char*)alloc((size_t)2 * 4 * 2048);
  unsigned*       smb    = (unsigned*)alloc((size_t)2 * 4 * 64 * 4);
  // dead-buffer aliases (stream order makes these safe):
  float* c0 = qkv;                                   // after k_rope, qkv f32 is dead
  float* c1 = qkv + (size_t)2 * 512 * 2048;
  float* c2 = c1 + (size_t)2 * 256 * 2048;
  unsigned short* attnout = hsb;                     // after gemm1, hsb is dead

  k_cvt_bf16 <<<32768, 256, 0, stream>>>(hidden, hsb);
  k_pack_wqkv<<<24576, 256, 0, stream>>>(Wq, Wk, Wv, wqkvb);
  k_cvt_bf16 <<<16384, 256, 0, stream>>>(Wo, wob);
  k_gemm_bf16<3072, 2048><<<dim3(24, 32), 256, 0, stream>>>(hsb, wqkvb, qkv);
  k_rope     <<<4096, 256, 0, stream>>>(qkv, qbb, kbb, vtb, convin);
  k_conv3    <<<8192, 256, 0, stream>>>(convin, cw0, cb0, c0, 512, 256, 1);
  k_conv3    <<<4096, 256, 0, stream>>>(c0, cw1, cb1, c1, 256, 128, 1);
  k_conv3    <<<2048, 256, 0, stream>>>(c1, cw2, cb2, c2, 128, 64, 1);
  k_proj_sel <<<64, 256, 0, stream>>>(c2, pw, pb, selb);
  k_selmask  <<<2, 256, 0, stream>>>(selb, smb);
  k_attn     <<<512, 256, 0, stream>>>(qbb, kbb, vtb, smb, attnout);
  k_gemm_bf16<2048, 2048><<<dim3(16, 32), 256, 0, stream>>>(attnout, wob, out);
}
